// MultiHeadAttention_32770600469129
// MI455X (gfx1250) — matrix operand order repaired and checked by reading, not compiled, not run on hardware
//
#include <hip/hip_runtime.h>

typedef __attribute__((ext_vector_type(16))) _Float16 v16h;
typedef __attribute__((ext_vector_type(8)))  _Float16 v8h;
typedef __attribute__((ext_vector_type(8)))  float    v8f;

#define WMMA_F16(a,b,c) \
  __builtin_amdgcn_wmma_f32_16x16x32_f16(false,(a),false,(b),(short)0,(c),false,false)

constexpr int BB = 2;      // batch
constexpr int S  = 2048;   // sequence
constexpr int E  = 1024;   // embed
constexpr int H  = 16;     // heads
constexpr int Dh = 64;     // head dim

__device__ inline v16h cat16(v8h lo, v8h hi) {
  return __builtin_shufflevector(lo, hi, 0,1,2,3,4,5,6,7,8,9,10,11,12,13,14,15);
}

// ---------------------------------------------------------------------------
// Prep: Wq/Wk/Wv [H,E,Dh] f32 -> transposed f16 [H,Dh,E];  Wo [E,E] f32 -> f16
// (H*E*Dh == E*E == 1M elements, exact grid coverage)
// ---------------------------------------------------------------------------
__global__ __launch_bounds__(256) void prep_kernel(
    const float* __restrict__ Wq, const float* __restrict__ Wk,
    const float* __restrict__ Wv, const float* __restrict__ Wo,
    _Float16* __restrict__ WqT, _Float16* __restrict__ WkT,
    _Float16* __restrict__ WvT, _Float16* __restrict__ Wo16) {
  int idx = blockIdx.x * 256 + threadIdx.x;        // < H*E*Dh
  int h   = idx / (E * Dh);
  int rem = idx % (E * Dh);
  int e   = rem / Dh;
  int d   = rem % Dh;
  size_t dst = (size_t)h * Dh * E + (size_t)d * E + e;
  WqT[dst] = (_Float16)Wq[idx];
  WkT[dst] = (_Float16)Wk[idx];
  WvT[dst] = (_Float16)Wv[idx];
  Wo16[idx] = (_Float16)Wo[idx];
}

// ---------------------------------------------------------------------------
// Projection: Y[b,h] = X[b] (SxE, f32) @ W[h] (ExDh via WT[h,Dh,E] f16)
// One wave computes a 16x64 tile. transposed=0: Y[B,H,S,Dh]; =1: Y[B,H,Dh,S]
// ---------------------------------------------------------------------------
__global__ __launch_bounds__(32) void proj_kernel(
    const float* __restrict__ X, const _Float16* __restrict__ WT,
    _Float16* __restrict__ Y, float scale, int transposed) {
  int tile = blockIdx.x;                 // B*H*(S/16)
  int mt   = tile % (S / 16);
  int bh   = tile / (S / 16);
  int h    = bh % H;
  int b    = bh / H;
  int lane = threadIdx.x;
  int col  = lane & 15;
  int half = lane >> 4;
  int base0 = half * 8;

  const float*    xrow = X + ((size_t)b * S + mt * 16 + col) * E;
  const _Float16* Wh   = WT + (size_t)h * Dh * E;

  v8f a0 = {}, a1 = {}, a2 = {}, a3 = {};
  for (int k0 = 0; k0 < E; k0 += 32) {
    v8h lo = __builtin_convertvector(*(const v8f*)(xrow + k0 + half * 16),      v8h);
    v8h hi = __builtin_convertvector(*(const v8f*)(xrow + k0 + half * 16 + 8), v8h);
    v16h a = cat16(lo, hi);
    int kb = k0 + half * 16;
    const _Float16* w0 = Wh + (size_t)col * E + kb;   // n = nt*16 + col rows
    v16h b0 = *(const v16h*)(w0);
    v16h b1 = *(const v16h*)(w0 + (size_t)16 * E);
    v16h b2 = *(const v16h*)(w0 + (size_t)32 * E);
    v16h b3 = *(const v16h*)(w0 + (size_t)48 * E);
    a0 = WMMA_F16(a, b0, a0);
    a1 = WMMA_F16(a, b1, a1);
    a2 = WMMA_F16(a, b2, a2);
    a3 = WMMA_F16(a, b3, a3);
  }

  size_t yb = (size_t)bh * S * Dh;
#pragma unroll
  for (int r = 0; r < 8; ++r) {
    int m = mt * 16 + base0 + r;
    _Float16 y0 = (_Float16)(a0[r] * scale);
    _Float16 y1 = (_Float16)(a1[r] * scale);
    _Float16 y2 = (_Float16)(a2[r] * scale);
    _Float16 y3 = (_Float16)(a3[r] * scale);
    if (!transposed) {
      _Float16* yr = Y + yb + (size_t)m * Dh;
      yr[col] = y0; yr[16 + col] = y1; yr[32 + col] = y2; yr[48 + col] = y3;
    } else {
      Y[yb + (size_t)(col)      * S + m] = y0;
      Y[yb + (size_t)(16 + col) * S + m] = y1;
      Y[yb + (size_t)(32 + col) * S + m] = y2;
      Y[yb + (size_t)(48 + col) * S + m] = y3;
    }
  }
}

// ---------------------------------------------------------------------------
// Flash attention: one wave per (b,h,16 query rows). Causal, online softmax.
// 64-key groups to amortize softmax VALU (shuffle tree, alpha rescale).
// qh,kh: [B,H,S,Dh] f16 (qh pre-scaled by 1/sqrt(Dh)); vT: [B,H,Dh,S] f16.
// concat: [B*S, E] f16, head h occupies columns [h*Dh, h*Dh+64).
// ---------------------------------------------------------------------------
__global__ __launch_bounds__(32) void flash_kernel(
    const _Float16* __restrict__ qh, const _Float16* __restrict__ kh,
    const _Float16* __restrict__ vT, _Float16* __restrict__ concat) {
  __shared__ __align__(32) _Float16 Pbuf[16 * 64];

  int tile = blockIdx.x;
  int qb   = tile % (S / 16);
  int bh   = tile / (S / 16);
  int h    = bh % H;
  int b    = bh / H;
  int lane = threadIdx.x;
  int col  = lane & 15;
  int half = lane >> 4;
  int base0 = half * 8;
  int q0   = qb * 16;

  const _Float16* Q = qh + (size_t)bh * S * Dh;
  const _Float16* K = kh + (size_t)bh * S * Dh;
  const _Float16* V = vT + (size_t)bh * Dh * S;

  const _Float16* qrow = Q + (size_t)(q0 + col) * Dh;
  v16h aq0 = cat16(*(const v8h*)(qrow + half * 16),      *(const v8h*)(qrow + half * 16 + 8));
  v16h aq1 = cat16(*(const v8h*)(qrow + 32 + half * 16), *(const v8h*)(qrow + 40 + half * 16));

  float m_i[8], l_i[8];
  v8f o0 = {}, o1 = {}, o2 = {}, o3 = {};
#pragma unroll
  for (int r = 0; r < 8; ++r) { m_i[r] = -3.0e38f; l_i[r] = 0.0f; }

  for (int g0 = 0; g0 <= q0 + 15; g0 += 64) {
    // ---- scores: four 16x16 tiles (keys g0+16t .. +15), K-dim = 64
    v8f s[4];
#pragma unroll
    for (int t = 0; t < 4; ++t) {
      v8f z = {};
      const _Float16* kr = K + (size_t)(g0 + 16 * t + col) * Dh;
      v16h bk0 = *(const v16h*)(kr + half * 16);
      v16h bk1 = *(const v16h*)(kr + 32 + half * 16);
      z = WMMA_F16(aq0, bk0, z);
      z = WMMA_F16(aq1, bk1, z);
      s[t] = z;
    }
    // ---- causal mask + online softmax update over 64 keys at once
#pragma unroll
    for (int r = 0; r < 8; ++r) {
      int row = q0 + base0 + r;
      float a0v = (g0 + col      > row) ? -3.0e38f : s[0][r];
      float a1v = (g0 + 16 + col > row) ? -3.0e38f : s[1][r];
      float a2v = (g0 + 32 + col > row) ? -3.0e38f : s[2][r];
      float a3v = (g0 + 48 + col > row) ? -3.0e38f : s[3][r];
      float rmax = fmaxf(fmaxf(a0v, a1v), fmaxf(a2v, a3v));
      rmax = fmaxf(rmax, __shfl_xor(rmax, 1, 32));
      rmax = fmaxf(rmax, __shfl_xor(rmax, 2, 32));
      rmax = fmaxf(rmax, __shfl_xor(rmax, 4, 32));
      rmax = fmaxf(rmax, __shfl_xor(rmax, 8, 32));
      float mnew  = fmaxf(m_i[r], rmax);
      float alpha = __expf(m_i[r] - mnew);
      float p0 = __expf(a0v - mnew);
      float p1 = __expf(a1v - mnew);
      float p2 = __expf(a2v - mnew);
      float p3 = __expf(a3v - mnew);
      float ps = (p0 + p1) + (p2 + p3);
      ps += __shfl_xor(ps, 1, 32);
      ps += __shfl_xor(ps, 2, 32);
      ps += __shfl_xor(ps, 4, 32);
      ps += __shfl_xor(ps, 8, 32);
      l_i[r] = l_i[r] * alpha + ps;
      m_i[r] = mnew;
      o0[r] *= alpha; o1[r] *= alpha; o2[r] *= alpha; o3[r] *= alpha;
      _Float16* prow = &Pbuf[(base0 + r) * 64];
      prow[col]      = (_Float16)p0;
      prow[16 + col] = (_Float16)p1;
      prow[32 + col] = (_Float16)p2;
      prow[48 + col] = (_Float16)p3;
    }
    __syncthreads();
    // ---- reload P as two 16x32 A-fragments (keys 0..31 and 32..63 of group)
    const _Float16* prow = &Pbuf[col * 64];
    v16h ap0 = cat16(*(const v8h*)(prow + half * 16),      *(const v8h*)(prow + half * 16 + 8));
    v16h ap1 = cat16(*(const v8h*)(prow + 32 + half * 16), *(const v8h*)(prow + 40 + half * 16));
    __syncthreads();
    // ---- O += P @ V  (B-fragments contiguous thanks to V transpose)
    {
      const _Float16* vb = V + (size_t)col * S + g0 + half * 16;
      v16h bv;
      bv = *(const v16h*)(vb);                     o0 = WMMA_F16(ap0, bv, o0);
      bv = *(const v16h*)(vb + 32);                o0 = WMMA_F16(ap1, bv, o0);
      bv = *(const v16h*)(vb + (size_t)16 * S);      o1 = WMMA_F16(ap0, bv, o1);
      bv = *(const v16h*)(vb + (size_t)16 * S + 32); o1 = WMMA_F16(ap1, bv, o1);
      bv = *(const v16h*)(vb + (size_t)32 * S);      o2 = WMMA_F16(ap0, bv, o2);
      bv = *(const v16h*)(vb + (size_t)32 * S + 32); o2 = WMMA_F16(ap1, bv, o2);
      bv = *(const v16h*)(vb + (size_t)48 * S);      o3 = WMMA_F16(ap0, bv, o3);
      bv = *(const v16h*)(vb + (size_t)48 * S + 32); o3 = WMMA_F16(ap1, bv, o3);
    }
  }

#pragma unroll
  for (int r = 0; r < 8; ++r) {
    float inv = 1.0f / l_i[r];
    size_t row = (size_t)b * S + q0 + base0 + r;
    _Float16* crow = concat + row * E + h * Dh;
    crow[col]      = (_Float16)(o0[r] * inv);
    crow[16 + col] = (_Float16)(o1[r] * inv);
    crow[32 + col] = (_Float16)(o2[r] * inv);
    crow[48 + col] = (_Float16)(o3[r] * inv);
  }
}

// ---------------------------------------------------------------------------
// Output projection: out = concat @ Wo^T + bo  (f16 WMMA, f32 out)
// One wave computes a 16x64 tile of the [B*S, E] output.
// ---------------------------------------------------------------------------
__global__ __launch_bounds__(32) void outproj_kernel(
    const _Float16* __restrict__ X, const _Float16* __restrict__ W,
    const float* __restrict__ bo, float* __restrict__ out) {
  int idx = blockIdx.x;                  // (B*S/16) * (E/64)
  int ng  = idx % (E / 64);
  int mt  = idx / (E / 64);
  int lane = threadIdx.x;
  int col  = lane & 15;
  int half = lane >> 4;
  int base0 = half * 8;

  const _Float16* arow = X + (size_t)(mt * 16 + col) * E;
  v8f a0 = {}, a1 = {}, a2 = {}, a3 = {};
  for (int k0 = 0; k0 < E; k0 += 32) {
    v16h a = cat16(*(const v8h*)(arow + k0 + half * 16),
                   *(const v8h*)(arow + k0 + half * 16 + 8));
    int kb = k0 + half * 16;
    const _Float16* w0 = W + (size_t)(ng * 64 + col) * E + kb;  // row n = ng*64+nt*16+col
    v16h b0 = *(const v16h*)(w0);
    v16h b1 = *(const v16h*)(w0 + (size_t)16 * E);
    v16h b2 = *(const v16h*)(w0 + (size_t)32 * E);
    v16h b3 = *(const v16h*)(w0 + (size_t)48 * E);
    a0 = WMMA_F16(a, b0, a0);
    a1 = WMMA_F16(a, b1, a1);
    a2 = WMMA_F16(a, b2, a2);
    a3 = WMMA_F16(a, b3, a3);
  }
#pragma unroll
  for (int r = 0; r < 8; ++r) {
    size_t row = (size_t)mt * 16 + base0 + r;
    float* orow = out + row * E + ng * 64;
    orow[col]      = a0[r] + bo[ng * 64 + col];
    orow[16 + col] = a1[r] + bo[ng * 64 + 16 + col];
    orow[32 + col] = a2[r] + bo[ng * 64 + 32 + col];
    orow[48 + col] = a3[r] + bo[ng * 64 + 48 + col];
  }
}

// ---------------------------------------------------------------------------
extern "C" void kernel_launch(void* const* d_in, const int* in_sizes, int n_in,
                              void* d_out, int out_size, void* d_ws, size_t ws_size,
                              hipStream_t stream) {
  (void)in_sizes; (void)n_in; (void)out_size; (void)ws_size;
  const float* q  = (const float*)d_in[0];
  const float* k  = (const float*)d_in[1];
  const float* v  = (const float*)d_in[2];
  const float* Wq = (const float*)d_in[3];
  const float* Wk = (const float*)d_in[4];
  const float* Wv = (const float*)d_in[5];
  const float* Wo = (const float*)d_in[6];
  const float* bo = (const float*)d_in[7];
  float* out = (float*)d_out;

  char* ws = (char*)d_ws;
  const size_t szWT     = (size_t)H * Dh * E * sizeof(_Float16);      // 2 MB
  const size_t szWo     = (size_t)E * E * sizeof(_Float16);           // 2 MB
  const size_t szHeads  = (size_t)BB * H * S * Dh * sizeof(_Float16); // 8 MB
  const size_t szConcat = (size_t)BB * S * E * sizeof(_Float16);      // 8 MB

  _Float16* WqT    = (_Float16*)ws;               ws += szWT;
  _Float16* WkT    = (_Float16*)ws;               ws += szWT;
  _Float16* WvT    = (_Float16*)ws;               ws += szWT;
  _Float16* Wo16   = (_Float16*)ws;               ws += szWo;
  _Float16* qh16   = (_Float16*)ws;               ws += szHeads;
  _Float16* kh16   = (_Float16*)ws;               ws += szHeads;
  _Float16* vhT16  = (_Float16*)ws;               ws += szHeads;
  _Float16* concat = (_Float16*)ws;               ws += szConcat;

  const float qscale = 0.125f;  // 1/sqrt(Dh)

  prep_kernel<<<dim3((H * E * Dh) / 256), dim3(256), 0, stream>>>(
      Wq, Wk, Wv, Wo, WqT, WkT, WvT, Wo16);

  const int projGrid = BB * H * (S / 16);   // 4096
  proj_kernel<<<dim3(projGrid), dim3(32), 0, stream>>>(q, WqT, qh16, qscale, 0);
  proj_kernel<<<dim3(projGrid), dim3(32), 0, stream>>>(k, WkT, kh16, 1.0f,   0);
  proj_kernel<<<dim3(projGrid), dim3(32), 0, stream>>>(v, WvT, vhT16, 1.0f,  1);

  flash_kernel<<<dim3(projGrid), dim3(32), 0, stream>>>(qh16, kh16, vhT16, concat);

  const int opGrid = (BB * S / 16) * (E / 64);  // 4096
  outproj_kernel<<<dim3(opGrid), dim3(32), 0, stream>>>(concat, Wo16, bo, out);
}
